// CasualAttention_30236569764370
// MI455X (gfx1250) — compile-verified, matrix operand order checked by reading, not run on hardware
//
#include <hip/hip_runtime.h>
#include <hip/hip_bf16.h>
#include <hip/hip_fp16.h>

typedef __attribute__((ext_vector_type(16))) _Float16 v16h;
typedef __attribute__((ext_vector_type(8)))  float    v8f;
typedef int v4i __attribute__((vector_size(16)));

#define B_SZ  4
#define SEQ   4096
#define DIN   512
#define DHEAD 64

// Async global->LDS path (gfx1250). Guarded so we fall back cleanly.
#if defined(__has_builtin)
#if __has_builtin(__builtin_amdgcn_global_load_async_to_lds_b128)
#define USE_ASYNC_LDS 1
#endif
#endif

// Builtin signature: (v4i AS(1)* gsrc, v4i AS(3)* ldsdst, imm offset, imm cpol)
#define AS_GLOBAL(p) ((__attribute__((address_space(1))) v4i*)(void*)(p))
#define AS_LDS(p)    ((__attribute__((address_space(3))) v4i*)(void*)(p))

__device__ __forceinline__ void async_wait_all() {
#if defined(__has_builtin)
#if __has_builtin(__builtin_amdgcn_s_wait_asynccnt)
    __builtin_amdgcn_s_wait_asynccnt(0);
#else
    asm volatile("s_wait_asynccnt 0x0" ::: "memory");
#endif
#else
    asm volatile("s_wait_asynccnt 0x0" ::: "memory");
#endif
}

// Fragment: 16 halves per lane = two 16-byte chunks (ISA 16-bit A 16x32 layout:
// lanes 0-15 hold K = 0..7 & 16..23, lanes 16-31 hold K = 8..15 & 24..31).
union Frag { v16h h; uint4 q[2]; };
union U8h  { uint4 q; _Float16 e[8]; };

__device__ __forceinline__ v8f wmma_f16(v16h a, v16h b, v8f c) {
    return __builtin_amdgcn_wmma_f32_16x16x32_f16(
        /*neg_a=*/false, a, /*neg_b=*/false, b,
        /*c_mod=*/(short)0, c, /*reuse_a=*/false, /*reuse_b=*/false);
}

// ---------------------------------------------------------------------------
// Kernel 0: convert + transpose weights: W[k,n] fp32 -> Wt[mat][n][k] f16
// ---------------------------------------------------------------------------
__global__ __launch_bounds__(256) void wconv_kernel(
    const float* __restrict__ Wq, const float* __restrict__ Wk,
    const float* __restrict__ Wv, _Float16* __restrict__ Wt) {
    int idx = blockIdx.x * 256 + threadIdx.x;
    const int per = DHEAD * DIN;
    if (idx >= 3 * per) return;
    int mat = idx / per;
    int rem = idx - mat * per;          // rem = n*DIN + k
    int n   = rem / DIN;
    int k   = rem - n * DIN;
    const float* W = (mat == 0) ? Wq : ((mat == 1) ? Wk : Wv);
    Wt[idx] = (_Float16)W[k * DHEAD + n];
}

// ---------------------------------------------------------------------------
// Kernel 1: QKV projection. 128 rows x 64 cols per block, 8 waves.
// ---------------------------------------------------------------------------
__global__ __launch_bounds__(256) void qkv_proj_kernel(
    const float* __restrict__ X, const _Float16* __restrict__ Wt,
    _Float16* __restrict__ Qh, _Float16* __restrict__ Kh,
    _Float16* __restrict__ Vh) {
    __shared__ alignas(16) _Float16 lds_x[128 * 32];

    const int lane = threadIdx.x & 31;
    const int w    = threadIdx.x >> 5;   // wave id 0..7, owns 16 rows
    const int g    = lane >> 4;          // half-wave group
    const int r    = lane & 15;

    const int gRowBase = blockIdx.x * 128;

    v8f acc[3][4];
    v8f zero = {};
#pragma unroll
    for (int m = 0; m < 3; ++m)
#pragma unroll
        for (int n = 0; n < 4; ++n) acc[m][n] = zero;

    for (int kc = 0; kc < DIN / 32; ++kc) {
        // Stage 128x32 chunk of X as f16 into LDS (1024 float4 loads).
#pragma unroll
        for (int i = 0; i < 4; ++i) {
            int idx = threadIdx.x + 256 * i;          // 0..1023
            int row = idx >> 3;
            int c4  = idx & 7;
            float4 xv = *(const float4*)(X + (size_t)(gRowBase + row) * DIN +
                                         kc * 32 + c4 * 4);
            _Float16* dst = lds_x + row * 32 + c4 * 4;
            dst[0] = (_Float16)xv.x; dst[1] = (_Float16)xv.y;
            dst[2] = (_Float16)xv.z; dst[3] = (_Float16)xv.w;
        }
        __syncthreads();

        // A fragment: this wave's 16-row strip of X-chunk.
        Frag fa;
        const _Float16* xr = lds_x + (w * 16 + r) * 32;
        fa.q[0] = *(const uint4*)(xr + g * 8);
        fa.q[1] = *(const uint4*)(xr + 16 + g * 8);

#pragma unroll
        for (int mat = 0; mat < 3; ++mat) {
#pragma unroll
            for (int n = 0; n < 4; ++n) {
                Frag fb;
                const _Float16* wr = Wt + (size_t)mat * DHEAD * DIN +
                                     (size_t)(n * 16 + r) * DIN + kc * 32;
                fb.q[0] = *(const uint4*)(wr + g * 8);
                fb.q[1] = *(const uint4*)(wr + 16 + g * 8);
                acc[mat][n] = wmma_f16(fa.h, fb.h, acc[mat][n]);
            }
        }
        __syncthreads();
    }

    // Epilogue: D layout M = v + 8*g (lanes), N = n*16 + r. Q scaled by 1/8.
#pragma unroll
    for (int mat = 0; mat < 3; ++mat) {
        _Float16* dst = (mat == 0) ? Qh : ((mat == 1) ? Kh : Vh);
        const float scale = (mat == 0) ? 0.125f : 1.0f;
#pragma unroll
        for (int n = 0; n < 4; ++n)
#pragma unroll
            for (int v = 0; v < 8; ++v) {
                int row = gRowBase + w * 16 + v + 8 * g;
                dst[(size_t)row * DHEAD + n * 16 + r] =
                    (_Float16)(acc[mat][n][v] * scale);
            }
    }
}

// ---------------------------------------------------------------------------
// Kernel 2: flash attention. Block = 128 query rows (8 waves x 16 rows),
// key tiles of 64, online softmax, f32 accum, f16 matrix ops.
// K/V tiles staged via gfx1250 async global->LDS copies when available.
// ---------------------------------------------------------------------------
__global__ __launch_bounds__(256) void attn_kernel(
    const _Float16* __restrict__ Qh, const _Float16* __restrict__ Kh,
    const _Float16* __restrict__ Vh, float* __restrict__ Out) {
    __shared__ alignas(16) _Float16 lds_k [64 * 64];   // [key][d]
    __shared__ alignas(16) _Float16 lds_v [64 * 64];   // [key][d] (async landing)
    __shared__ alignas(16) _Float16 lds_vt[64 * 64];   // [d][key] (transposed)
    __shared__ alignas(16) _Float16 lds_p [8 * 16 * 64]; // per-wave P strips

    const int lane = threadIdx.x & 31;
    const int w    = threadIdx.x >> 5;
    const int g    = lane >> 4;
    const int r    = lane & 15;

    const int qtiles = SEQ / 128;                // 32
    const int b      = blockIdx.x / qtiles;
    const int qt     = blockIdx.x - b * qtiles;
    const int qBase  = qt * 128;

    const _Float16* Qb = Qh + (size_t)b * SEQ * DHEAD;
    const _Float16* Kb = Kh + (size_t)b * SEQ * DHEAD;
    const _Float16* Vb = Vh + (size_t)b * SEQ * DHEAD;

    // Load this wave's Q fragments once (16 rows x 64 dims, 2 K-chunks).
    Frag fq[2];
    const _Float16* qr = Qb + (size_t)(qBase + w * 16 + r) * DHEAD;
#pragma unroll
    for (int kc = 0; kc < 2; ++kc) {
        fq[kc].q[0] = *(const uint4*)(qr + kc * 32 + g * 8);
        fq[kc].q[1] = *(const uint4*)(qr + kc * 32 + 16 + g * 8);
    }

    float m_row[8], l_row[8];
    v8f acc_o[4];
    v8f zero = {};
#pragma unroll
    for (int v = 0; v < 8; ++v) { m_row[v] = -1e30f; l_row[v] = 0.0f; }
#pragma unroll
    for (int n = 0; n < 4; ++n) acc_o[n] = zero;

    const int ktmax = 2 * qt + 1;     // key tiles up to & including diagonal
    for (int kt = 0; kt <= ktmax; ++kt) {
        const int kb = kt * 64;

#ifdef USE_ASYNC_LDS
        // Async DMA-style staging: global -> LDS without VGPR round-trip.
#pragma unroll
        for (int i = 0; i < 2; ++i) {
            int u   = threadIdx.x + 256 * i;      // 0..511
            int key = u >> 3;
            int c8  = u & 7;
            __builtin_amdgcn_global_load_async_to_lds_b128(
                AS_GLOBAL(Kb + (size_t)(kb + key) * DHEAD + c8 * 8),
                AS_LDS(lds_k + key * DHEAD + c8 * 8), 0, 0);
            __builtin_amdgcn_global_load_async_to_lds_b128(
                AS_GLOBAL(Vb + (size_t)(kb + key) * DHEAD + c8 * 8),
                AS_LDS(lds_v + key * DHEAD + c8 * 8), 0, 0);
        }
        if (kt < ktmax)
            __builtin_prefetch(Kb + (size_t)(kb + 64) * DHEAD, 0, 1);
        async_wait_all();
        __syncthreads();

        // Transpose V inside LDS for contiguous PV B-fragments.
#pragma unroll
        for (int i = 0; i < 2; ++i) {
            int u   = threadIdx.x + 256 * i;
            int key = u >> 3;
            int c8  = u & 7;
            U8h vv;
            vv.q = *(const uint4*)(lds_v + key * DHEAD + c8 * 8);
#pragma unroll
            for (int j = 0; j < 8; ++j)
                lds_vt[(c8 * 8 + j) * 64 + key] = vv.e[j];
        }
        __syncthreads();
#else
        // Fallback: synchronous staging (K row-major; V transposed on the fly).
#pragma unroll
        for (int i = 0; i < 2; ++i) {
            int u   = threadIdx.x + 256 * i;      // 0..511
            int key = u >> 3;
            int c8  = u & 7;
            *(uint4*)(lds_k + key * DHEAD + c8 * 8) =
                *(const uint4*)(Kb + (size_t)(kb + key) * DHEAD + c8 * 8);
            U8h vv;
            vv.q = *(const uint4*)(Vb + (size_t)(kb + key) * DHEAD + c8 * 8);
#pragma unroll
            for (int j = 0; j < 8; ++j)
                lds_vt[(c8 * 8 + j) * 64 + key] = vv.e[j];
        }
        if (kt < ktmax)
            __builtin_prefetch(Kb + (size_t)(kb + 64) * DHEAD, 0, 1);
        __syncthreads();
#endif

        // S = (Q*scale) @ K^T for this wave's 16 rows x 64 keys.
        v8f s_acc[4];
#pragma unroll
        for (int n = 0; n < 4; ++n) s_acc[n] = zero;
#pragma unroll
        for (int n = 0; n < 4; ++n)
#pragma unroll
            for (int kc = 0; kc < 2; ++kc) {
                Frag fk;
                const _Float16* kr = lds_k + (n * 16 + r) * DHEAD + kc * 32;
                fk.q[0] = *(const uint4*)(kr + g * 8);
                fk.q[1] = *(const uint4*)(kr + 16 + g * 8);
                s_acc[n] = wmma_f16(fq[kc].h, fk.h, s_acc[n]);
            }

        // Causal mask + online softmax (row M = v + 8*g, col N = n*16 + r).
#pragma unroll
        for (int v = 0; v < 8; ++v) {
            const int qrow = qBase + w * 16 + v + 8 * g;
            float rowmax = -1e30f;
#pragma unroll
            for (int n = 0; n < 4; ++n) {
                int key  = kb + n * 16 + r;
                float sv = s_acc[n][v];
                if (key > qrow) sv = -1e30f;
                s_acc[n][v] = sv;
                rowmax = fmaxf(rowmax, sv);
            }
#pragma unroll
            for (int msk = 1; msk < 16; msk <<= 1)
                rowmax = fmaxf(rowmax, __shfl_xor(rowmax, msk, 32));

            float mnew  = fmaxf(m_row[v], rowmax);
            float alpha = __expf(m_row[v] - mnew);
            m_row[v]    = mnew;

            float rs = 0.0f;
#pragma unroll
            for (int n = 0; n < 4; ++n) {
                float p = __expf(s_acc[n][v] - mnew);
                s_acc[n][v] = p;
                rs += p;
            }
#pragma unroll
            for (int msk = 1; msk < 16; msk <<= 1)
                rs += __shfl_xor(rs, msk, 32);

            l_row[v] = l_row[v] * alpha + rs;
#pragma unroll
            for (int nn = 0; nn < 4; ++nn) acc_o[nn][v] *= alpha;
        }

        // P -> f16 in this wave's private LDS strip, re-read as A fragments.
        _Float16* pw = lds_p + w * 16 * 64;
#pragma unroll
        for (int n = 0; n < 4; ++n)
#pragma unroll
            for (int v = 0; v < 8; ++v)
                pw[(v + 8 * g) * 64 + n * 16 + r] = (_Float16)s_acc[n][v];

        // O += P @ V  (V^T staged in LDS -> contiguous B fragments).
#pragma unroll
        for (int kc = 0; kc < 2; ++kc) {
            Frag fp;
            const _Float16* pr = pw + r * 64 + kc * 32;
            fp.q[0] = *(const uint4*)(pr + g * 8);
            fp.q[1] = *(const uint4*)(pr + 16 + g * 8);
#pragma unroll
            for (int nn = 0; nn < 4; ++nn) {
                Frag fv;
                const _Float16* vr = lds_vt + (nn * 16 + r) * 64 + kc * 32;
                fv.q[0] = *(const uint4*)(vr + g * 8);
                fv.q[1] = *(const uint4*)(vr + 16 + g * 8);
                acc_o[nn] = wmma_f16(fp.h, fv.h, acc_o[nn]);
            }
        }
        __syncthreads();   // before next tile overwrites lds_k / lds_v / lds_vt
    }

    // Epilogue: divide by running sum, write fp32 output.
#pragma unroll
    for (int nn = 0; nn < 4; ++nn)
#pragma unroll
        for (int v = 0; v < 8; ++v) {
            int row = qBase + w * 16 + v + 8 * g;
            float o = acc_o[nn][v] / l_row[v];
            Out[((size_t)b * SEQ + row) * DHEAD + nn * 16 + r] = o;
        }
}

// ---------------------------------------------------------------------------
extern "C" void kernel_launch(void* const* d_in, const int* in_sizes, int n_in,
                              void* d_out, int out_size, void* d_ws, size_t ws_size,
                              hipStream_t stream) {
    const float* X  = (const float*)d_in[0];
    const float* Wq = (const float*)d_in[1];
    const float* Wk = (const float*)d_in[2];
    const float* Wv = (const float*)d_in[3];
    float* Out = (float*)d_out;

    char* ws = (char*)d_ws;
    const size_t wt_bytes  = (size_t)3 * DHEAD * DIN * sizeof(_Float16);   // 192 KB
    const size_t qkv_bytes = (size_t)B_SZ * SEQ * DHEAD * sizeof(_Float16); // 2 MB each
    _Float16* Wt = (_Float16*)(ws);
    _Float16* Qh = (_Float16*)(ws + wt_bytes);
    _Float16* Kh = (_Float16*)(ws + wt_bytes + qkv_bytes);
    _Float16* Vh = (_Float16*)(ws + wt_bytes + 2 * qkv_bytes);

    // 1) weight transpose/convert: 3*64*512 = 98304 elements
    wconv_kernel<<<(3 * DHEAD * DIN + 255) / 256, 256, 0, stream>>>(Wq, Wk, Wv, Wt);

    // 2) QKV projection: one block per 128 rows
    qkv_proj_kernel<<<(B_SZ * SEQ) / 128, 256, 0, stream>>>(X, Wt, Qh, Kh, Vh);

    // 3) flash attention: one block per (batch, 128-row query tile)
    attn_kernel<<<B_SZ * (SEQ / 128), 256, 0, stream>>>(Qh, Kh, Vh, Out);
}
